// RelativeMultiHeadAttentionModule_47631187313307
// MI455X (gfx1250) — matrix operand order repaired and checked by reading, not compiled, not run on hardware
//
#include <hip/hip_runtime.h>

// Problem constants (from reference)
#define B_  4
#define S_  1024
#define D_  512
#define H_  8
#define DH_ 64
#define M_  4096   // B*S

typedef __attribute__((ext_vector_type(16))) __bf16 v16bf;
typedef __attribute__((ext_vector_type(8)))  float  v8f;
typedef __attribute__((ext_vector_type(2)))  float  f32x2;   // clang ext-vector (nontemporal-OK)

union BfVec {
    unsigned int   d[8];
    unsigned short u[16];
    v16bf          v;
};

__device__ __forceinline__ unsigned short f32_to_bf16(float f) {
    unsigned int x = __float_as_uint(f);
    x += 0x7FFFu + ((x >> 16) & 1u);   // round-to-nearest-even
    return (unsigned short)(x >> 16);
}

__device__ __forceinline__ v8f wmma_bf16(const BfVec& a, const BfVec& b, v8f c) {
    return __builtin_amdgcn_wmma_f32_16x16x32_bf16(
        /*neg_a=*/false, a.v, /*neg_b=*/false, b.v,
        /*c_mod=*/(short)0, c, /*reuse_a=*/false, /*reuse_b=*/false);
}

// 16-bit A-matrix 16x32 layout (ISA 7.12.2): K offset of VGPR pair vg
__device__ __forceinline__ int a_koff(int vg, int half) {
    return (vg < 4 ? 2 * vg : 16 + 2 * (vg - 4)) + half * 8;
}
// 16-bit B-matrix 32x16 layout: K offset of VGPR pair vg
__device__ __forceinline__ int b_koff(int vg, int half) {
    return (vg < 4 ? 2 * vg : 16 + 2 * (vg - 4)) + half * 8;
}

// ---------------------------------------------------------------------------
// Convert a 512x512 f32 weight (stored [K,N]) into transposed bf16 Wt[N,K].
// ---------------------------------------------------------------------------
__global__ void wt_convert_kernel(const float* __restrict__ W,
                                  unsigned short* __restrict__ Wt) {
    const int idx = blockIdx.x * 256 + threadIdx.x;   // 0 .. 512*512-1
    const int k = idx >> 9, n = idx & 511;
    Wt[(size_t)n * D_ + k] = f32_to_bf16(W[idx]);
}

// ---------------------------------------------------------------------------
// Projection GEMM: Y = X[M,512] @ W[512,512] (+bias), W as bf16 Wt[N,K].
// Block = 128 thr (4 waves); wave computes 16(M) x 64(N).
// Template flags specialize the epilogue (no runtime branches):
//   HBI: add bias[n];  HF: f32 out [M,512];  HA/HB: bf16 per-head outputs
//   EA/EB: add per-(h,d) extra bias;  TR: [B,H,DH,S] layout (for V)
// ---------------------------------------------------------------------------
template <bool HBI, bool HF, bool HA, bool EA, bool HB, bool EB, bool TR>
__global__ void proj_gemm_kernel(const float* __restrict__ X,
                                 const unsigned short* __restrict__ Wt,
                                 const float* __restrict__ bias,
                                 float* __restrict__ outF,
                                 unsigned short* __restrict__ outA,
                                 const float* __restrict__ extraA,
                                 unsigned short* __restrict__ outB,
                                 const float* __restrict__ extraB) {
    const int lane = threadIdx.x & 31;
    const int wave = threadIdx.x >> 5;      // 0..3
    const int half = lane >> 4;
    const int lr   = lane & 15;
    const int row0  = blockIdx.x * 64 + wave * 16;
    const int ncol0 = blockIdx.y * 64;

    const float* xrow = X + (size_t)(row0 + lr) * D_;

    v8f z = {};
    v8f acc[4] = {z, z, z, z};
    for (int k0 = 0; k0 < D_; k0 += 32) {
        __builtin_prefetch(xrow + k0 + 32, 0, 1);
        BfVec a, b[4];
#pragma unroll
        for (int j = 0; j < 4; ++j) {
            const unsigned short* wrow = Wt + (size_t)(ncol0 + j * 16 + lr) * D_ + k0;
#pragma unroll
            for (int vg = 0; vg < 8; ++vg)
                b[j].d[vg] = *(const unsigned int*)(wrow + b_koff(vg, half));
        }
#pragma unroll
        for (int vg = 0; vg < 8; ++vg) {
            const int ka = a_koff(vg, half) + k0;
            f32x2 f = *(const f32x2*)(xrow + ka);
            a.u[2 * vg]     = f32_to_bf16(f.x);
            a.u[2 * vg + 1] = f32_to_bf16(f.y);
        }
#pragma unroll
        for (int j = 0; j < 4; ++j)
            acc[j] = wmma_bf16(a, b[j], acc[j]);
    }

#pragma unroll
    for (int j = 0; j < 4; ++j) {
        const int   n  = ncol0 + j * 16 + lr;
        const float bv = HBI ? bias[n] : 0.0f;
        const int   h  = n >> 6, dd = n & 63;
        const int   hd = h * DH_ + dd;
        const float ea = EA ? extraA[hd] : 0.0f;
        const float eb = EB ? extraB[hd] : 0.0f;
#pragma unroll
        for (int vg = 0; vg < 8; ++vg) {
            const int   m   = row0 + vg + 8 * half;
            const float val = acc[j][vg] + bv;
            if (HF) outF[(size_t)m * D_ + n] = val;
            const int bb = m >> 10, ss = m & (S_ - 1);
            const size_t idx = TR
                ? (((size_t)bb * H_ + h) * DH_ + dd) * S_ + ss
                : (((size_t)bb * H_ + h) * S_ + ss) * DH_ + dd;
            if (HA) outA[idx] = f32_to_bf16(val + ea);
            if (HB) outB[idx] = f32_to_bf16(val + eb);
        }
    }
}

// ---------------------------------------------------------------------------
// Scores per (b,h): content = (q+u)·k^T -> d_out attn area; pos = (q+v)·p^T -> ws
// Wave computes 16(s) x 64(t) for BOTH matrices; loads batched before WMMAs.
// ---------------------------------------------------------------------------
__global__ void score_kernel(const unsigned short* __restrict__ Qc,
                             const unsigned short* __restrict__ Qp,
                             const unsigned short* __restrict__ Kb,
                             const unsigned short* __restrict__ Pb,
                             float* __restrict__ content,
                             float* __restrict__ posw) {
    const int lane = threadIdx.x & 31;
    const int wave = threadIdx.x >> 5;      // 0..7
    const int half = lane >> 4;
    const int lr   = lane & 15;
    const int s0 = (blockIdx.x * 8 + wave) * 16;
    const int t0 = blockIdx.y * 64;
    const int bh = blockIdx.z;
    const size_t off = (size_t)bh * S_ * DH_;
    const unsigned short* qc = Qc + off + (size_t)(s0 + lr) * DH_;
    const unsigned short* qp = Qp + off + (size_t)(s0 + lr) * DH_;
    const unsigned short* kb = Kb + off;
    const unsigned short* pb = Pb + off;

    v8f z = {};
    v8f accC[4] = {z, z, z, z};
    v8f accP[4] = {z, z, z, z};
#pragma unroll
    for (int k0 = 0; k0 < DH_; k0 += 32) {
        BfVec ac, ap, bk[4], bp[4];
#pragma unroll
        for (int vg = 0; vg < 8; ++vg) {
            const int ka = a_koff(vg, half) + k0;
            ac.d[vg] = *(const unsigned int*)(qc + ka);
            ap.d[vg] = *(const unsigned int*)(qp + ka);
        }
#pragma unroll
        for (int j = 0; j < 4; ++j) {
            const size_t trow = (size_t)(t0 + j * 16 + lr) * DH_ + k0;
#pragma unroll
            for (int vg = 0; vg < 8; ++vg) {
                const int kk = b_koff(vg, half);   // K-dim = head dim d
                bk[j].d[vg] = *(const unsigned int*)(kb + trow + kk);
                bp[j].d[vg] = *(const unsigned int*)(pb + trow + kk);
            }
        }
#pragma unroll
        for (int j = 0; j < 4; ++j) {
            accC[j] = wmma_bf16(ac, bk[j], accC[j]);
            accP[j] = wmma_bf16(ap, bp[j], accP[j]);
        }
    }

    const size_t cbase = (size_t)bh * S_ * S_;
#pragma unroll
    for (int j = 0; j < 4; ++j) {
#pragma unroll
        for (int vg = 0; vg < 8; ++vg) {
            const int srow = s0 + vg + 8 * half;
            const size_t idx = cbase + (size_t)srow * S_ + t0 + j * 16 + lr;
            __builtin_nontemporal_store(accC[j][vg], content + idx);
            __builtin_nontemporal_store(accP[j][vg], posw + idx);
        }
    }
}

// ---------------------------------------------------------------------------
// Exact _relative_shift + scale + softmax, one block per score row, in place.
// shifted[k,l]: t=(k+1)*S+l; i=t/(S+1); j=t%(S+1); val=(j==0)?0:pos[i,j-1]
// ---------------------------------------------------------------------------
__global__ void shift_softmax_kernel(float* __restrict__ attn,
                                     const float* __restrict__ posw) {
    const int bh   = blockIdx.x >> 10;        // / S_
    const int krow = blockIdx.x & (S_ - 1);
    const int tid  = threadIdx.x;
    __shared__ float row[S_];
    __shared__ float red[256];
    float* arow = attn + ((size_t)bh * S_ + krow) * S_;
    const float* pb = posw + (size_t)bh * S_ * S_;
    const float scale = 0.04419417382415922f;  // 1/sqrt(512)

    float lmax = -3.0e38f;
    for (int l = tid; l < S_; l += 256) {
        const int t = (krow + 1) * S_ + l;
        const int i = t / (S_ + 1);
        const int j = t - i * (S_ + 1);
        const float pv = (j == 0) ? 0.0f
                                  : __builtin_nontemporal_load(pb + (size_t)i * S_ + (j - 1));
        const float v = (__builtin_nontemporal_load(arow + l) + pv) * scale;
        row[l] = v;
        lmax = fmaxf(lmax, v);
    }
    red[tid] = lmax; __syncthreads();
    for (int s = 128; s > 0; s >>= 1) {
        if (tid < s) red[tid] = fmaxf(red[tid], red[tid + s]);
        __syncthreads();
    }
    const float mx = red[0];
    __syncthreads();
    float lsum = 0.0f;
    for (int l = tid; l < S_; l += 256) {
        const float e = __expf(row[l] - mx);
        row[l] = e;
        lsum += e;
    }
    red[tid] = lsum; __syncthreads();
    for (int s = 128; s > 0; s >>= 1) {
        if (tid < s) red[tid] += red[tid + s];
        __syncthreads();
    }
    const float inv = 1.0f / red[0];
    for (int l = tid; l < S_; l += 256)
        __builtin_nontemporal_store(row[l] * inv, arow + l);
}

// ---------------------------------------------------------------------------
// context[b,s,h*64+d] = sum_t attn[b,h,s,t] * V[b,h,t,d]
// V stored transposed bf16 [B,H,DH,S]; wave computes 16 rows x all 64 dims.
// ---------------------------------------------------------------------------
__global__ void context_kernel(const float* __restrict__ attn,
                               const unsigned short* __restrict__ Vt,
                               float* __restrict__ ctx) {   // [B,S,D] f32
    const int lane = threadIdx.x & 31;
    const int wave = threadIdx.x >> 5;      // 0..7
    const int half = lane >> 4;
    const int lr   = lane & 15;
    const int s0 = (blockIdx.x * 8 + wave) * 16;
    const int bh = blockIdx.y;
    const int bb = bh >> 3, h = bh & 7;
    const float*          arow = attn + (size_t)bh * S_ * S_ + (size_t)(s0 + lr) * S_;
    const unsigned short* vt   = Vt   + (size_t)bh * DH_ * S_;

    v8f z = {};
    v8f acc[4] = {z, z, z, z};
    for (int t0 = 0; t0 < S_; t0 += 32) {
        __builtin_prefetch(arow + t0 + 32, 0, 1);
        BfVec a, b[4];
#pragma unroll
        for (int j = 0; j < 4; ++j) {
            const unsigned short* vrow = vt + (size_t)(j * 16 + lr) * S_ + t0;
#pragma unroll
            for (int vg = 0; vg < 8; ++vg)
                b[j].d[vg] = *(const unsigned int*)(vrow + b_koff(vg, half));  // K-dim = t
        }
#pragma unroll
        for (int vg = 0; vg < 8; ++vg) {
            const int ka = a_koff(vg, half) + t0;
            f32x2 f = __builtin_nontemporal_load((const f32x2*)(arow + ka));
            a.u[2 * vg]     = f32_to_bf16(f.x);
            a.u[2 * vg + 1] = f32_to_bf16(f.y);
        }
#pragma unroll
        for (int j = 0; j < 4; ++j)
            acc[j] = wmma_bf16(a, b[j], acc[j]);
    }
#pragma unroll
    for (int j = 0; j < 4; ++j) {
#pragma unroll
        for (int vg = 0; vg < 8; ++vg) {
            const int s = s0 + vg + 8 * half;
            ctx[((size_t)bb * S_ + s) * D_ + h * DH_ + j * 16 + lr] = acc[j][vg];
        }
    }
}

// ---------------------------------------------------------------------------
extern "C" void kernel_launch(void* const* d_in, const int* in_sizes, int n_in,
                              void* d_out, int out_size, void* d_ws, size_t ws_size,
                              hipStream_t stream) {
    const float* query = (const float*)d_in[0];
    const float* key   = (const float*)d_in[1];
    const float* value = (const float*)d_in[2];
    const float* pose  = (const float*)d_in[3];
    const float* Wq = (const float*)d_in[4];
    const float* bq = (const float*)d_in[5];
    const float* Wk = (const float*)d_in[6];
    const float* bk = (const float*)d_in[7];
    const float* Wv = (const float*)d_in[8];
    const float* bv = (const float*)d_in[9];
    const float* Wp = (const float*)d_in[10];
    const float* Wo = (const float*)d_in[11];
    const float* bo = (const float*)d_in[12];
    const float* ub = (const float*)d_in[13];
    const float* vb = (const float*)d_in[14];

    // Workspace layout:
    //   5 x bf16[B*H*S*DH]  (Qc,Qp,Kb,Pb,Vt)                = 20 MB
    //   5 x bf16[512*512]   (transposed bf16 weights)       = 2.5 MB
    //   ctx  f32 [B,S,D]                                    = 8 MB
    //   posw f32 [B,H,S,S]                                  = 128 MB
    const size_t QKV_E = (size_t)B_ * H_ * S_ * DH_;   // 2,097,152
    const size_t W_E   = (size_t)D_ * D_;              // 262,144
    unsigned short* Qc  = (unsigned short*)d_ws;
    unsigned short* Qp  = Qc + QKV_E;
    unsigned short* Kb  = Qp + QKV_E;
    unsigned short* Pb  = Kb + QKV_E;
    unsigned short* Vt  = Pb + QKV_E;
    unsigned short* Wtq = Vt + QKV_E;
    unsigned short* Wtk = Wtq + W_E;
    unsigned short* Wtv = Wtk + W_E;
    unsigned short* Wtp = Wtv + W_E;
    unsigned short* Wto = Wtp + W_E;
    float* ctx  = (float*)(Wto + W_E);
    float* posw = ctx + (size_t)B_ * S_ * D_;

    float* outp = (float*)d_out;                       // [B,S,D]
    float* attn = outp + (size_t)B_ * S_ * D_;         // [B,H,S,S]

    // Pre-convert weights to transposed bf16 (tiny; stays resident in L2)
    const dim3 wg(W_E / 256);
    wt_convert_kernel<<<wg, 256, 0, stream>>>(Wq, Wtq);
    wt_convert_kernel<<<wg, 256, 0, stream>>>(Wk, Wtk);
    wt_convert_kernel<<<wg, 256, 0, stream>>>(Wv, Wtv);
    wt_convert_kernel<<<wg, 256, 0, stream>>>(Wp, Wtp);
    wt_convert_kernel<<<wg, 256, 0, stream>>>(Wo, Wto);

    const dim3 pg(M_ / 64, D_ / 64);                   // (64, 8), 128 thr

    // Q: bf16 copies with +u (content) and +v (position) biases
    proj_gemm_kernel<true, false, true, true, true, true, false>
        <<<pg, 128, 0, stream>>>(query, Wtq, bq, nullptr, Qc, ub, Qp, vb);
    // K, V (transposed), P
    proj_gemm_kernel<true, false, true, false, false, false, false>
        <<<pg, 128, 0, stream>>>(key,   Wtk, bk, nullptr, Kb, nullptr, nullptr, nullptr);
    proj_gemm_kernel<true, false, true, false, false, false, true>
        <<<pg, 128, 0, stream>>>(value, Wtv, bv, nullptr, Vt, nullptr, nullptr, nullptr);
    proj_gemm_kernel<false, false, true, false, false, false, false>
        <<<pg, 128, 0, stream>>>(pose,  Wtp, nullptr, nullptr, Pb, nullptr, nullptr, nullptr);

    // Content (-> d_out attn area) and position (-> ws) scores
    score_kernel<<<dim3(S_ / 128, S_ / 64, B_ * H_), 256, 0, stream>>>(Qc, Qp, Kb, Pb, attn, posw);

    // Relative shift + scale + softmax, in place on the attn area
    shift_softmax_kernel<<<dim3(B_ * H_ * S_), 256, 0, stream>>>(attn, posw);

    // context = attn @ V, then out = ctx @ Wo + bo
    context_kernel<<<dim3(S_ / 128, B_ * H_), 256, 0, stream>>>(attn, Vt, ctx);
    proj_gemm_kernel<true, true, false, false, false, false, false>
        <<<pg, 128, 0, stream>>>(ctx, Wto, bo, outp, nullptr, nullptr, nullptr, nullptr);
}
